// cox_loss_v2_85263690760315
// MI455X (gfx1250) — compile-verified, matrix operand order checked by reading, not run on hardware
//
#include <hip/hip_runtime.h>

typedef __attribute__((ext_vector_type(16))) _Float16 v16h;
typedef __attribute__((ext_vector_type(8)))  _Float16 v8h;
typedef __attribute__((ext_vector_type(8)))  float    v8f;
typedef __attribute__((ext_vector_type(4)))  float    v4f;
typedef __attribute__((ext_vector_type(4)))  int      v4i;

// Address-space-qualified pointer-to-int4 types for the async LDS builtin
// (HIP diagnostic showed param type as "int4 __device__ *" == addrspace(1) int4*).
typedef __attribute__((address_space(1))) v4i* as1_v4i_p;
typedef __attribute__((address_space(3))) v4i* as3_v4i_p;

#define COX_N 8192

// gfx1250 async global->LDS path, hedged so compilation succeeds on any toolchain.
#if defined(__gfx1250__) && __has_builtin(__builtin_amdgcn_global_load_async_to_lds_b128) && \
    __has_builtin(__builtin_amdgcn_s_wait_asynccnt)
#define COX_ASYNC_LDS 1
#else
#define COX_ASYNC_LDS 0
#endif

// Kernel 1: exp(theta) -> f16 (only exp is quantized; mask compares stay f32-exact)
__global__ __launch_bounds__(256) void cox_exp_kernel(const float* __restrict__ hazard,
                                                      _Float16* __restrict__ exp_h) {
  int i = blockIdx.x * 256 + threadIdx.x;
  if (i < COX_N) {
    __builtin_prefetch(&hazard[i], 0, 0);   // global_prefetch_b8
    exp_h[i] = (_Float16)__expf(hazard[i]);
  }
}

// Kernel 2: risk_sum[i] = sum_k exp(theta_k) * (surv_k >= surv_i) via v_wmma_f32_16x16x32_f16.
// A[i][k] = masked exp_f16 built on the fly from LDS, B = ones => D[:,0] = risk_sum tile.
__global__ __launch_bounds__(256) void cox_wmma_risk_kernel(const float* __restrict__ y_true,
                                                            const _Float16* __restrict__ exp_h,
                                                            float* __restrict__ risk_out) {
  __shared__ float    s_surv[COX_N];   // 32 KB
  __shared__ _Float16 s_exp[COX_N];    // 16 KB  (48 KB of 320 KB/WGP)
  const int t = threadIdx.x;

#if COX_ASYNC_LDS
  // Contiguous 16 KB of exp_f16: stage with async b128 (16 B/lane, 4 passes).
  for (int j = t * 8; j < COX_N; j += 256 * 8) {
    __builtin_amdgcn_global_load_async_to_lds_b128(
        (as1_v4i_p)&exp_h[j],
        (as3_v4i_p)&s_exp[j], 0, 0);
  }
#else
  for (int j = t; j < COX_N; j += 256) s_exp[j] = exp_h[j];
#endif
  // survtime = y_true[:,0] is a stride-2 de-interleave: regular loads.
  for (int j = t; j < COX_N; j += 256) s_surv[j] = y_true[2 * j];
#if COX_ASYNC_LDS
  __builtin_amdgcn_s_wait_asynccnt(0);
#endif
  __syncthreads();

  const int wave   = t >> 5;                    // 8 waves per block
  const int lane   = t & 31;
  const int tile   = blockIdx.x * 8 + wave;     // 512 row-tiles total
  const int i_base = tile * 16;
  const float st_i = s_surv[i_base + (lane & 15)];  // this lane's row threshold
  const int hi     = (lane >> 4) & 1;           // lane group selects K+8 half

  v16h b_ones;
#pragma unroll
  for (int e = 0; e < 16; ++e) b_ones[e] = (_Float16)1.0f;

  v8f c = {};
  const _Float16 h0 = (_Float16)0.0f;

#pragma unroll 4
  for (int k0 = 0; k0 < COX_N; k0 += 32) {
    // ISA 16-bit A 16x32 layout: lanes 0-15 hold K = {0..7, 16..23}, lanes 16-31 = +8.
    const int kb = k0 + (hi << 3);
    v4f s0 = *(const v4f*)&s_surv[kb];          // K = kb+0..3
    v4f s1 = *(const v4f*)&s_surv[kb + 4];      // K = kb+4..7
    v4f s2 = *(const v4f*)&s_surv[kb + 16];     // K = kb+16..19
    v4f s3 = *(const v4f*)&s_surv[kb + 20];     // K = kb+20..23
    v8h e0 = *(const v8h*)&s_exp[kb];           // exp for K = kb+0..7
    v8h e1 = *(const v8h*)&s_exp[kb + 16];      // exp for K = kb+16..23
    v16h a;
#pragma unroll
    for (int e = 0; e < 4; ++e) {
      a[e]      = (s0[e] >= st_i) ? e0[e]     : h0;
      a[e + 4]  = (s1[e] >= st_i) ? e0[e + 4] : h0;
      a[e + 8]  = (s2[e] >= st_i) ? e1[e]     : h0;
      a[e + 12] = (s3[e] >= st_i) ? e1[e + 4] : h0;
    }
    // 8 args: (neg_a, A, neg_b, B, c_mod, C, reuse_a, reuse_b)
    c = __builtin_amdgcn_wmma_f32_16x16x32_f16(false, a, false, b_ones,
                                               (short)0, c, false, false);
  }

  // C/D layout: VGPR r, lane 0 -> (M=r, N=0); lane 16 -> (M=r+8, N=0). Column 0 = risk_sum.
  if (lane == 0) {
#pragma unroll
    for (int r = 0; r < 8; ++r) risk_out[i_base + r] = c[r];
  } else if (lane == 16) {
#pragma unroll
    for (int r = 0; r < 8; ++r) risk_out[i_base + 8 + r] = c[r];
  }
}

// Kernel 3: loss = -(1/N) * sum_i censor_i * (theta_i - log(risk_sum_i)); single block, deterministic.
__global__ __launch_bounds__(1024) void cox_loss_reduce_kernel(const float* __restrict__ y_true,
                                                               const float* __restrict__ hazard,
                                                               const float* __restrict__ risk,
                                                               float* __restrict__ out) {
  __shared__ float red[1024];
  float acc = 0.0f;
  for (int i = threadIdx.x; i < COX_N; i += 1024) {
    float censor = (y_true[2 * i + 1] != 0.0f) ? 1.0f : 0.0f;
    float theta  = hazard[i];
    acc += censor * (theta - __logf(risk[i]));
  }
  red[threadIdx.x] = acc;
  __syncthreads();
  for (int s = 512; s > 0; s >>= 1) {
    if (threadIdx.x < s) red[threadIdx.x] += red[threadIdx.x + s];
    __syncthreads();
  }
  if (threadIdx.x == 0) out[0] = -red[0] / (float)COX_N;
}

extern "C" void kernel_launch(void* const* d_in, const int* in_sizes, int n_in,
                              void* d_out, int out_size, void* d_ws, size_t ws_size,
                              hipStream_t stream) {
  const float* y_true = (const float*)d_in[0];   // (N,2): [survtime, censor]
  const float* hazard = (const float*)d_in[1];   // (N,1) flat
  _Float16* exp_h = (_Float16*)d_ws;                                  // 16 KB
  float*    risk  = (float*)((char*)d_ws + COX_N * sizeof(_Float16)); // next 32 KB
  float*    out   = (float*)d_out;

  cox_exp_kernel<<<COX_N / 256, 256, 0, stream>>>(hazard, exp_h);
  cox_wmma_risk_kernel<<<COX_N / 128, 256, 0, stream>>>(y_true, exp_h, risk);
  cox_loss_reduce_kernel<<<1, 1024, 0, stream>>>(y_true, hazard, risk, out);
}
